// SpikingAntennalLobe_42880953483370
// MI455X (gfx1250) — compile-verified, matrix operand order checked by reading, not run on hardware
//
#include <hip/hip_runtime.h>
#include <stdint.h>

// ---------------- problem dimensions ----------------
#define BATCH   256
#define N_ORN   2048
#define N_LN    512
#define N_PN    512
#define N_STEPS 200
#define KDIM    (N_ORN + N_LN)   // 2560 : concat [s_o | s_ln] (weight K)
#define NDIM    (N_LN + N_PN)    // 1024 : concat [inj_l | inj_p]

// ---------------- model constants (SI) ----------------
#define DT            1e-3f
#define R_M           1e9f
#define V_REST        (-0.06f)
#define V_RESET       (-0.06f)
#define V_MIN         (-0.08f)
#define TAU_REF       2e-3f
#define ALPHA_SYN     0.8187307531f     // exp(-1e-3/5e-3)
#define I_NOISE_STD   5e-12f
#define V_NOISE_STD   5e-4f
#define TH_JITTER_STD 2e-4f
#define SYN_NOISE_STD 0.15f
#define ORN_NOISE_STD 0.05f
#define INV_TEMP      1000.0f

// noise salts
#define SALT_ORN_MULT 13u
#define SALT_ORN      10u
#define SALT_LN       20u
#define SALT_PN       30u
#define SALT_SYN      40u

typedef __bf16 bf16x8 __attribute__((ext_vector_type(8)));
typedef __bf16 v16bf  __attribute__((ext_vector_type(16)));
typedef float  v8f    __attribute__((ext_vector_type(8)));

// ---------------- deterministic RNG (hash -> approx normal) ----------------
__device__ __forceinline__ uint32_t hashu(uint32_t x) {
  x ^= x >> 16; x *= 0x7feb352du;
  x ^= x >> 15; x *= 0x846ca68bu;
  x ^= x >> 16; return x;
}
__device__ __forceinline__ float nrand(uint32_t salt, uint32_t step, uint32_t idx) {
  uint32_t base = salt * 0x9E3779B9u ^ step * 0x85EBCA6Bu ^ idx * 0xC2B2AE35u;
  float s = 0.0f;
#pragma unroll
  for (uint32_t j = 0; j < 4; ++j)
    s += (float)(hashu(base + j * 0x27D4EB2Fu) >> 8) * (1.0f / 16777216.0f);
  return (s - 2.0f) * 1.7320508f;   // mean 0, var ~1
}

// ---------------- LIF step (exact exponential integration + soft spike) ----
__device__ __forceinline__ void lif_step(float I, float& v, float& refr,
                                         float log_tau, float v_th,
                                         uint32_t saltbase, uint32_t step,
                                         uint32_t idx, float& spike) {
  float rn = fmaxf(refr - DT, 0.0f);
  bool can_fire = (rn <= 0.0f);
  I += nrand(saltbase + 0u, step, idx) * I_NOISE_STD;
  float v_inf = V_REST + R_M * I;
  float alpha = __expf(-DT / __expf(log_tau));
  float vn = v_inf + (v - v_inf) * alpha;
  vn += nrand(saltbase + 1u, step, idx) * V_NOISE_STD;
  vn = fmaxf(vn, V_MIN);
  vn = can_fire ? vn : V_RESET;
  float th = v_th + nrand(saltbase + 2u, step, idx) * TH_JITTER_STD;
  float sp = 1.0f / (1.0f + __expf(-(vn - th) * INV_TEMP));
  sp = can_fire ? sp : 0.0f;
  if (sp > 0.5f) { vn = V_RESET; rn = TAU_REF; }
  v = vn; refr = rn; spike = sp;
}

// ---------------- utility fills / reductions ----------------
__global__ void fill_u32_kernel(uint32_t* p, int n, uint32_t v) {
  int i = blockIdx.x * blockDim.x + threadIdx.x;
  if (i < n) p[i] = v;
}
__global__ void fill_f32_kernel(float* p, int n, float v) {
  int i = blockIdx.x * blockDim.x + threadIdx.x;
  if (i < n) p[i] = v;
}
__global__ void reduce_max_i32_kernel(const int* c, int n, int* out) {
  int i = blockIdx.x * blockDim.x + threadIdx.x;
  if (i < n) atomicMax(out, c[i]);
}

// ---------------- fused weight build: Bt[n][k], bf16 ----------------
// n<512 -> LN target; n>=512 -> PN target. k<2048 -> ORN source; else LN source.
__global__ void build_weights_kernel(__bf16* Bt,
                                     const int* cop, const int* col,
                                     const int* clp, const int* cll,
                                     const float* ls_op, const float* ls_ol,
                                     const float* ls_lp, const float* ls_ll,
                                     const int* maxc) {
  int i = blockIdx.x * blockDim.x + threadIdx.x;
  const int total = NDIM * KDIM;
  if (i >= total) return;
  int n = i / KDIM;
  int k = i - n * KDIM;
  float w;
  if (n < N_LN) {
    if (k < N_ORN) {
      int m = maxc[1]; float c = (float)col[(size_t)k * N_LN + n];
      float norm = (m > 0) ? c / fmaxf((float)m, 1.0f) : c;
      w = __expf(ls_ol[0]) * norm;
    } else {
      int m = maxc[3]; float c = (float)cll[(size_t)(k - N_ORN) * N_LN + n];
      float norm = (m > 0) ? c / fmaxf((float)m, 1.0f) : c;
      w = -__expf(ls_ll[0]) * norm;
    }
  } else {
    int j = n - N_LN;
    if (k < N_ORN) {
      int m = maxc[0]; float c = (float)cop[(size_t)k * N_PN + j];
      float norm = (m > 0) ? c / fmaxf((float)m, 1.0f) : c;
      w = __expf(ls_op[0]) * norm;
    } else {
      int m = maxc[2]; float c = (float)clp[(size_t)(k - N_ORN) * N_PN + j];
      float norm = (m > 0) ? c / fmaxf((float)m, 1.0f) : c;
      w = -__expf(ls_lp[0]) * norm;
    }
  }
  Bt[(size_t)n * KDIM + k] = (__bf16)w;
}

// ---------------- ORN LIF step: writes bf16 spikes into A_orn [256][2048] --
__global__ void orn_step_kernel(const float* __restrict__ orn_in,
                                float* v, float* r,
                                const float* __restrict__ vth,
                                const float* __restrict__ log_tau,
                                __bf16* __restrict__ A_orn, int step) {
  int idx = blockIdx.x * blockDim.x + threadIdx.x;
  if (idx >= BATCH * N_ORN) return;
  int i = idx & (N_ORN - 1);
  float Iin = orn_in[idx] * (1.0f + nrand(SALT_ORN_MULT, (uint32_t)step, (uint32_t)idx) * ORN_NOISE_STD);
  float vv = v[idx], rr = r[idx], sp;
  lif_step(Iin, vv, rr, log_tau[0], vth[i], SALT_ORN, (uint32_t)step, (uint32_t)idx, sp);
  v[idx] = vv; r[idx] = rr;
  A_orn[idx] = (__bf16)sp;
}

// ---- fragment load helper: two aligned 16B loads per lane, per ISA layout --
__device__ __forceinline__ v16bf load_frag(const __bf16* row, int k0) {
  bf16x8 lo = *(const bf16x8*)(row + k0);
  bf16x8 hi = *(const bf16x8*)(row + k0 + 16);
  return __builtin_shufflevector(lo, hi, 0,1,2,3,4,5,6,7,8,9,10,11,12,13,14,15);
}

// ---------------- fused GEMM + synaptic update + LN/PN LIF ----------------
// C = [s_o | s_ln_prev] (256x2560 bf16) x Bt^T (2560x1024 bf16), f32 acc.
// One wave computes a 16(M) x 32(N) slab (2 WMMA tiles sharing the A frag);
// 8 waves/block -> 16 x 256 strip; grid (4, 16) = 64 blocks.
// Epilogue: I = I*alpha + inj*(1+noise), then the full LN/PN LIF update.
// LN spikes go to S_next (ping-pong buffer, read by the NEXT step's K-tail).
__global__ void __launch_bounds__(256)
syn_gemm_lnpn_kernel(const __bf16* __restrict__ A_orn,   // [256][2048]
                     const __bf16* __restrict__ S_prev,  // [256][512]
                     __bf16* __restrict__ S_next,        // [256][512]
                     const __bf16* __restrict__ Bt,      // [1024][2560]
                     float* __restrict__ Icur,
                     float* __restrict__ v, float* __restrict__ r,
                     const float* __restrict__ vth_ln,
                     const float* __restrict__ vth_pn,
                     const float* __restrict__ lt_ln,
                     const float* __restrict__ lt_pn,
                     float* __restrict__ out, int step) {
  const int lane = threadIdx.x & 31;
  const int wave = threadIdx.x >> 5;
  const int half = lane >> 4;       // K-group selector
  const int lrow = lane & 15;       // M (for A) / N (for B) row in tile
  const int m0 = blockIdx.y * 16;
  const int n0 = blockIdx.x * 256 + wave * 32;

  const __bf16* arow  = A_orn + (size_t)(m0 + lrow) * N_ORN + half * 8;
  const __bf16* srow  = S_prev + (size_t)(m0 + lrow) * N_LN + half * 8;
  const __bf16* brow0 = Bt + (size_t)(n0 + lrow) * KDIM + half * 8;
  const __bf16* brow1 = Bt + (size_t)(n0 + 16 + lrow) * KDIM + half * 8;

  v8f acc0 = {}, acc1 = {};

  // K part 1: ORN spikes (K = 2048)
#pragma unroll 4
  for (int k0 = 0; k0 < N_ORN; k0 += 32) {
    v16bf av = load_frag(arow, k0);
    v16bf b0 = load_frag(brow0, k0);
    v16bf b1 = load_frag(brow1, k0);
    __builtin_prefetch(brow0 + k0 + 128, 0, 1);
    __builtin_prefetch(brow1 + k0 + 128, 0, 1);
    acc0 = __builtin_amdgcn_wmma_f32_16x16x32_bf16(false, av, false, b0,
                                                   (short)0, acc0, false, false);
    acc1 = __builtin_amdgcn_wmma_f32_16x16x32_bf16(false, av, false, b1,
                                                   (short)0, acc1, false, false);
  }
  // K part 2: previous LN spikes (K = 512), weight rows offset by 2048
#pragma unroll 4
  for (int k0 = 0; k0 < N_LN; k0 += 32) {
    v16bf av = load_frag(srow, k0);
    v16bf b0 = load_frag(brow0 + N_ORN, k0);
    v16bf b1 = load_frag(brow1 + N_ORN, k0);
    acc0 = __builtin_amdgcn_wmma_f32_16x16x32_bf16(false, av, false, b0,
                                                   (short)0, acc0, false, false);
    acc1 = __builtin_amdgcn_wmma_f32_16x16x32_bf16(false, av, false, b1,
                                                   (short)0, acc1, false, false);
  }

  // epilogue: synaptic current update + LN/PN LIF, per (b, n)
#pragma unroll
  for (int t = 0; t < 2; ++t) {
    v8f acc = t ? acc1 : acc0;
#pragma unroll
    for (int e = 0; e < 8; ++e) {
      int b = m0 + half * 8 + e;
      int n = n0 + t * 16 + lrow;
      size_t gi = (size_t)b * NDIM + n;
      float noise = nrand(SALT_SYN, (uint32_t)step, (uint32_t)gi);
      float Inew = Icur[gi] * ALPHA_SYN + acc[e] * (1.0f + noise * SYN_NOISE_STD);
      Icur[gi] = Inew;

      bool is_ln = (n < N_LN);
      float vth = is_ln ? vth_ln[n] : vth_pn[n - N_LN];
      float lt  = is_ln ? lt_ln[0] : lt_pn[0];
      float vv = v[gi], rr = r[gi], sp;
      lif_step(Inew, vv, rr, lt, vth, is_ln ? SALT_LN : SALT_PN,
               (uint32_t)step, (uint32_t)gi, sp);
      v[gi] = vv; r[gi] = rr;
      if (is_ln) {
        S_next[(size_t)b * N_LN + n] = (__bf16)sp;
      } else {
        out[(size_t)step * (BATCH * N_PN) + (size_t)b * N_PN + (n - N_LN)] = sp;
      }
    }
  }
}

// ---------------- workspace layout ----------------
static constexpr size_t OFF_AORN = 0;
static constexpr size_t SZ_AORN  = (size_t)BATCH * N_ORN * 2;  // bf16 ORN spikes
static constexpr size_t OFF_S0   = OFF_AORN + SZ_AORN;
static constexpr size_t SZ_S     = (size_t)BATCH * N_LN * 2;   // bf16 LN spikes
static constexpr size_t OFF_S1   = OFF_S0 + SZ_S;
static constexpr size_t OFF_BT   = OFF_S1 + SZ_S;
static constexpr size_t SZ_BT    = (size_t)NDIM * KDIM * 2;    // bf16 weights^T
static constexpr size_t OFF_VO   = OFF_BT + SZ_BT;
static constexpr size_t SZ_VO    = (size_t)BATCH * N_ORN * 4;
static constexpr size_t OFF_RO   = OFF_VO + SZ_VO;
static constexpr size_t OFF_VLP  = OFF_RO + SZ_VO;
static constexpr size_t SZ_VLP   = (size_t)BATCH * NDIM * 4;
static constexpr size_t OFF_RLP  = OFF_VLP + SZ_VLP;
static constexpr size_t OFF_I    = OFF_RLP + SZ_VLP;
static constexpr size_t OFF_MAX  = OFF_I + SZ_VLP;             // 4 ints

extern "C" void kernel_launch(void* const* d_in, const int* in_sizes, int n_in,
                              void* d_out, int out_size, void* d_ws, size_t ws_size,
                              hipStream_t stream) {
  (void)in_sizes; (void)n_in; (void)out_size; (void)ws_size;

  const float* orn_input  = (const float*)d_in[0];
  const float* lt_orn     = (const float*)d_in[1];
  const float* lt_ln      = (const float*)d_in[2];
  const float* lt_pn      = (const float*)d_in[3];
  const float* vth_orn    = (const float*)d_in[4];
  const float* vth_ln     = (const float*)d_in[5];
  const float* vth_pn     = (const float*)d_in[6];
  const float* ls_op      = (const float*)d_in[7];
  const float* ls_ol      = (const float*)d_in[8];
  const float* ls_lp      = (const float*)d_in[9];
  const float* ls_ll      = (const float*)d_in[10];
  const int*   counts_op  = (const int*)d_in[11];
  const int*   counts_ol  = (const int*)d_in[12];
  const int*   counts_lp  = (const int*)d_in[13];
  const int*   counts_ll  = (const int*)d_in[14];

  unsigned char* base = (unsigned char*)d_ws;
  __bf16* A_orn = (__bf16*)(base + OFF_AORN);
  __bf16* S[2]  = { (__bf16*)(base + OFF_S0), (__bf16*)(base + OFF_S1) };
  __bf16* Bt    = (__bf16*)(base + OFF_BT);
  float*  v_o   = (float*)(base + OFF_VO);
  float*  r_o   = (float*)(base + OFF_RO);
  float*  v_lp  = (float*)(base + OFF_VLP);
  float*  r_lp  = (float*)(base + OFF_RLP);
  float*  Icur  = (float*)(base + OFF_I);
  int*    maxc  = (int*)(base + OFF_MAX);
  float*  out   = (float*)d_out;

  const int T = 256;
  auto blks = [](int n, int t) { return (n + t - 1) / t; };

  // ---- weight preprocessing (runs every launch: deterministic) ----
  fill_u32_kernel<<<1, 32, 0, stream>>>((uint32_t*)maxc, 4, 0u);
  reduce_max_i32_kernel<<<blks(N_ORN * N_PN, T), T, 0, stream>>>(counts_op, N_ORN * N_PN, maxc + 0);
  reduce_max_i32_kernel<<<blks(N_ORN * N_LN, T), T, 0, stream>>>(counts_ol, N_ORN * N_LN, maxc + 1);
  reduce_max_i32_kernel<<<blks(N_LN * N_PN, T), T, 0, stream>>>(counts_lp, N_LN * N_PN, maxc + 2);
  reduce_max_i32_kernel<<<blks(N_LN * N_LN, T), T, 0, stream>>>(counts_ll, N_LN * N_LN, maxc + 3);
  build_weights_kernel<<<blks(NDIM * KDIM, T), T, 0, stream>>>(
      Bt, counts_op, counts_ol, counts_lp, counts_ll, ls_op, ls_ol, ls_lp, ls_ll, maxc);

  // ---- state init (every launch; harness does not re-initialize ws) ----
  fill_f32_kernel<<<blks(BATCH * N_ORN, T), T, 0, stream>>>(v_o, BATCH * N_ORN, V_RESET);
  fill_f32_kernel<<<blks(BATCH * N_ORN, T), T, 0, stream>>>(r_o, BATCH * N_ORN, 0.0f);
  fill_f32_kernel<<<blks(BATCH * NDIM, T), T, 0, stream>>>(v_lp, BATCH * NDIM, V_RESET);
  fill_f32_kernel<<<blks(BATCH * NDIM, T), T, 0, stream>>>(r_lp, BATCH * NDIM, 0.0f);
  fill_f32_kernel<<<blks(BATCH * NDIM, T), T, 0, stream>>>(Icur, BATCH * NDIM, 0.0f);
  fill_u32_kernel<<<blks((int)(SZ_S / 2), T), T, 0, stream>>>((uint32_t*)S[0], (int)(SZ_S / 2), 0u);
  fill_u32_kernel<<<blks((int)(SZ_S / 2), T), T, 0, stream>>>((uint32_t*)S[1], (int)(SZ_S / 2), 0u);

  // ---- 200 recurrent steps: 2 kernels per step ----
  dim3 gemm_grid(NDIM / 256, BATCH / 16);   // 4 x 16 blocks, 8 waves each
  for (int t = 0; t < N_STEPS; ++t) {
    orn_step_kernel<<<blks(BATCH * N_ORN, T), T, 0, stream>>>(
        orn_input, v_o, r_o, vth_orn, lt_orn, A_orn, t);
    syn_gemm_lnpn_kernel<<<gemm_grid, 256, 0, stream>>>(
        A_orn, S[t & 1], S[(t + 1) & 1], Bt, Icur, v_lp, r_lp,
        vth_ln, vth_pn, lt_ln, lt_pn, out, t);
  }
}
